// NewLLM_70841190580748
// MI455X (gfx1250) — matrix operand order repaired and checked by reading, not compiled, not run on hardware
//
#include <hip/hip_runtime.h>
#include <hip/hip_bf16.h>

// ---------------------------------------------------------------------------
// NewLLM for MI455X (gfx1250, wave32).
// Phase 0: embed+LN; Wo f32->bf16 WITH transpose to [N][K] (one-time, so the
//          hot GEMM needs no LDS transpose and B staging is all b128).
// Phase 1: persistent 16-block scan; W1/W2/Wu column slices pinned in LDS
//          (176KB of 320KB/WGP) across all 512 steps; 3 global barriers/step
//          (LN recomputed redundantly per block to avoid a 4th).
// Phase 2: H(bf16) @ Wo(bf16) -> f32 logits via v_wmma_f32_16x16x32_bf16.
//          Block tile 128x64 / wave tile 16x64 (4x v8f accumulators); k-loop
//          FULLY UNROLLED so accumulators stay pinned -> in-place wmma
//          accumulate, no mov/nop filler. NT stores for the 524MB logits
//          stream (22us floor @ 23.3TB/s); Wo(32MB bf16)+H(4MB) stay resident
//          in the 192MB L2.
// ---------------------------------------------------------------------------

typedef __attribute__((ext_vector_type(16))) __bf16 v16bf;
typedef __attribute__((ext_vector_type(8)))  float  v8f;

static constexpr int kB    = 8;
static constexpr int kS    = 512;
static constexpr int kE    = 256;
static constexpr int kCTX  = 256;
static constexpr int kHID  = 512;
static constexpr int kV    = 32000;
static constexpr int kKTOT = 512;      // E + CTX
static constexpr int kNBLK = 16;       // phase-1 persistent blocks
static constexpr float kEPS = 1e-5f;

// workspace layout (bytes), ~41MB total
static constexpr size_t OFF_BAR    = 0;                                   // 2 ints
static constexpr size_t OFF_CTXRAW = 1024;                                // [8][256] f32
static constexpr size_t OFF_HBUF   = OFF_CTXRAW + (size_t)kB*kCTX*4;      // [8][512] f32
static constexpr size_t OFF_H2     = OFF_HBUF   + (size_t)kB*kHID*4;      // [8][512] f32
static constexpr size_t OFF_TOK    = OFF_H2     + (size_t)kB*kHID*4;      // [512][8][256] f32
static constexpr size_t OFF_HBF    = OFF_TOK    + (size_t)kS*kB*kE*4;     // [4096][512] bf16
static constexpr size_t OFF_WOBF   = OFF_HBF    + (size_t)kS*kB*kHID*2;   // [32000][512] bf16 (transposed!)

__device__ __forceinline__ unsigned short bf16r(float f) {
    unsigned int u = __float_as_uint(f);
    return (unsigned short)((u + 0x7FFFu + ((u >> 16) & 1u)) >> 16);  // RNE
}

// ---------------------------------------------------------------------------
// Phase 0a: tok = LN(emb_table[ids]) * g + b   -> ws tok[t][b][c]
// Also re-initializes the global barrier words each call (deterministic).
// ---------------------------------------------------------------------------
__global__ void k_embed(const int* __restrict__ ids, const float* __restrict__ emb,
                        const float* __restrict__ g, const float* __restrict__ bb,
                        float* __restrict__ tok, int* __restrict__ bar) {
    const int blk = blockIdx.x;          // b*512 + t
    const int b = blk >> 9, t = blk & 511;
    const int c = threadIdx.x;           // 0..255
    if (blk == 0 && c < 8) bar[c] = 0;
    const int id = ids[blk];
    const float v = emb[(size_t)id * kE + c];
    float s1 = v, s2 = v * v;
    for (int m = 16; m >= 1; m >>= 1) { s1 += __shfl_xor(s1, m, 32); s2 += __shfl_xor(s2, m, 32); }
    __shared__ float p1[8], p2[8];
    const int w = c >> 5, l = c & 31;
    if (l == 0) { p1[w] = s1; p2[w] = s2; }
    __syncthreads();
    float t1 = 0.f, t2 = 0.f;
    for (int i = 0; i < 8; ++i) { t1 += p1[i]; t2 += p2[i]; }
    const float mean = t1 * (1.f / kE);
    const float var  = t2 * (1.f / kE) - mean * mean;
    const float rs   = rsqrtf(var + kEPS);
    tok[((size_t)t * kB + b) * kE + c] = (v - mean) * rs * g[c] + bb[c];
}

// ---------------------------------------------------------------------------
// Phase 0b: Wo [K=512][N=32000] f32 -> Wob [N][K] bf16 (LDS-tiled transpose).
// One-time ~100MB of traffic; afterwards the GEMM's B-staging is pure b128.
// ---------------------------------------------------------------------------
__global__ void k_cvt_wo(const float* __restrict__ wo, unsigned short* __restrict__ wob) {
    __shared__ unsigned short tile[32][33];
    const int n0 = blockIdx.x * 32;          // 1000 tiles along N
    const int k0 = blockIdx.y * 32;          // 16 tiles along K
    const int tx = threadIdx.x & 31, ty = threadIdx.x >> 5;   // 32 x 8
    #pragma unroll
    for (int i = 0; i < 4; ++i) {
        const int k = ty + i * 8;
        tile[k][tx] = bf16r(wo[(size_t)(k0 + k) * kV + n0 + tx]);   // coalesced along N
    }
    __syncthreads();
    #pragma unroll
    for (int i = 0; i < 4; ++i) {
        const int n = ty + i * 8;
        wob[(size_t)(n0 + n) * kHID + k0 + tx] = tile[tx][n];       // coalesced along K
    }
}

// ---------------------------------------------------------------------------
// sense-reversing global barrier across kNBLK resident blocks
// ---------------------------------------------------------------------------
__device__ __forceinline__ void gbar(int* bar) {
    __syncthreads();
    if (threadIdx.x == 0) {
        volatile int* vgen = bar + 1;
        const int g = *vgen;            // read generation BEFORE arriving
        __threadfence();
        if (atomicAdd(bar, 1) == kNBLK - 1) {
            *(volatile int*)bar = 0;
            __threadfence();
            atomicAdd(bar + 1, 1);
        } else {
            while (*vgen == g) __builtin_amdgcn_s_sleep(2);
        }
        __threadfence();
    }
    __syncthreads();
}

// LN+clip of one ctx row: lane l owns cols [8l, 8l+8). Returns the 8 values.
struct LN8 { float4 a, b; };
__device__ __forceinline__ LN8 ln_row(const float* __restrict__ src, int l,
                                      float4 cg0, float4 cg1, float4 cb0, float4 cb1) {
    const float4 r0 = *(const float4*)(src + l * 8);
    const float4 r1 = *(const float4*)(src + l * 8 + 4);
    float s1 = r0.x + r0.y + r0.z + r0.w + r1.x + r1.y + r1.z + r1.w;
    float s2 = r0.x*r0.x + r0.y*r0.y + r0.z*r0.z + r0.w*r0.w
             + r1.x*r1.x + r1.y*r1.y + r1.z*r1.z + r1.w*r1.w;
    for (int m = 16; m >= 1; m >>= 1) { s1 += __shfl_xor(s1, m, 32); s2 += __shfl_xor(s2, m, 32); }
    const float mean = s1 * (1.f / kCTX);
    const float var  = s2 * (1.f / kCTX) - mean * mean;
    const float rs   = rsqrtf(var + kEPS);
    auto f = [&](float v, float g, float b) {
        const float o = (v - mean) * rs * g + b;
        return fminf(10.f, fmaxf(-10.f, o));
    };
    LN8 o;
    o.a = make_float4(f(r0.x, cg0.x, cb0.x), f(r0.y, cg0.y, cb0.y),
                      f(r0.z, cg0.z, cb0.z), f(r0.w, cg0.w, cb0.w));
    o.b = make_float4(f(r1.x, cg1.x, cb1.x), f(r1.y, cg1.y, cb1.y),
                      f(r1.z, cg1.z, cb1.z), f(r1.w, cg1.w, cb1.w));
    return o;
}

// ---------------------------------------------------------------------------
// Phase 1: sequential scan. Block `blk` owns HID columns [blk*32, blk*32+32)
// of W1/W2 and CTX columns [blk*16, blk*16+16) of Wu, pinned in LDS.
// 3 global barriers per step; LN recomputed redundantly by every block.
// ---------------------------------------------------------------------------
__global__ void __launch_bounds__(256, 1) k_scan(
    const float* __restrict__ W1, const float* __restrict__ b1,
    const float* __restrict__ W2, const float* __restrict__ b2,
    const float* __restrict__ Wu, const float* __restrict__ bu,
    const float* __restrict__ cng, const float* __restrict__ cnb,
    const float* __restrict__ tok,
    float* __restrict__ ctx_raw,
    float* __restrict__ hbuf, float* __restrict__ h2buf,
    unsigned short* __restrict__ Hbf,
    float* __restrict__ traj, int* __restrict__ bar) {
    extern __shared__ float lds[];
    float* w1s = lds;                    // [512][32]
    float* w2s = w1s + kKTOT * 32;       // [512][32]
    float* wus = w2s + kHID * 32;        // [512][16]
    float* xs  = wus + kHID * 16;        // [8][512] staging
    const int j = threadIdx.x, blk = blockIdx.x;

    for (int idx = j; idx < kKTOT * 32; idx += 256) {
        const int r = idx >> 5, c = idx & 31;
        w1s[idx] = W1[(size_t)r * kHID + blk * 32 + c];
        w2s[idx] = W2[(size_t)r * kHID + blk * 32 + c];
    }
    for (int idx = j; idx < kHID * 16; idx += 256) {
        const int r = idx >> 4, c = idx & 15;
        wus[idx] = Wu[(size_t)r * kCTX + blk * 16 + c];
    }
    const int hb = j >> 5, hc = j & 31;            // wave -> batch row, lane -> col
    const float b1v = b1[blk * 32 + hc];
    const float b2v = b2[blk * 32 + hc];
    const float buv = (j < kB * 16) ? bu[blk * 16 + (j & 15)] : 0.f;
    // per-lane LN params: lane l owns ctx cols [8l, 8l+8) in every wave
    const int l = j & 31;
    const float4 cg0 = *(const float4*)(cng + l * 8);
    const float4 cg1 = *(const float4*)(cng + l * 8 + 4);
    const float4 cb0 = *(const float4*)(cnb + l * 8);
    const float4 cb1 = *(const float4*)(cnb + l * 8 + 4);
    __syncthreads();

    for (int t = 0; t < kS; ++t) {
        // ---- stage x = [tok_t | LN(ctx_raw)] ----
        for (int idx = j; idx < kB * kE / 4; idx += 256) {         // 512 float4s
            const int b = idx >> 6, c4 = idx & 63;
            *(float4*)(xs + b * kKTOT + c4 * 4) =
                *(const float4*)(tok + ((size_t)t * kB + b) * kE + c4 * 4);
        }
        if (t == 0) {
            const float4 z = make_float4(0.f, 0.f, 0.f, 0.f);
            for (int idx = j; idx < kB * kCTX / 4; idx += 256) {
                const int b = idx >> 6, c4 = idx & 63;
                *(float4*)(xs + b * kKTOT + kE + c4 * 4) = z;
            }
        } else {
            const LN8 o = ln_row(ctx_raw + hb * kCTX, l, cg0, cg1, cb0, cb1);
            *(float4*)(xs + hb * kKTOT + kE + l * 8)     = o.a;
            *(float4*)(xs + hb * kKTOT + kE + l * 8 + 4) = o.b;
            if (blk == 0) {   // traj for step t-1 (new_ctx of previous step)
                *(float4*)(traj + ((size_t)hb * kS + (t - 1)) * kCTX + l * 8)     = o.a;
                *(float4*)(traj + ((size_t)hb * kS + (t - 1)) * kCTX + l * 8 + 4) = o.b;
            }
        }
        __syncthreads();
        // ---- h1 = relu(x @ W1slice + b1) ----
        {
            float acc = b1v;
            const float* xr = xs + hb * kKTOT;
            const float* wc = w1s + hc;
            #pragma unroll 4
            for (int k = 0; k < kKTOT; k += 4) {
                const float4 xv = *(const float4*)(xr + k);
                acc = fmaf(xv.x, wc[(k + 0) * 32], acc);
                acc = fmaf(xv.y, wc[(k + 1) * 32], acc);
                acc = fmaf(xv.z, wc[(k + 2) * 32], acc);
                acc = fmaf(xv.w, wc[(k + 3) * 32], acc);
            }
            hbuf[hb * kHID + blk * 32 + hc] = fmaxf(acc, 0.f);
        }
        gbar(bar);
        for (int idx = j; idx < kB * kHID / 4; idx += 256)
            ((float4*)xs)[idx] = ((const float4*)hbuf)[idx];
        __syncthreads();
        // ---- h2 = relu(h1 @ W2slice + b2); export bf16 row for phase 2 ----
        {
            float acc = b2v;
            const float* xr = xs + hb * kHID;
            const float* wc = w2s + hc;
            #pragma unroll 4
            for (int k = 0; k < kHID; k += 4) {
                const float4 xv = *(const float4*)(xr + k);
                acc = fmaf(xv.x, wc[(k + 0) * 32], acc);
                acc = fmaf(xv.y, wc[(k + 1) * 32], acc);
                acc = fmaf(xv.z, wc[(k + 2) * 32], acc);
                acc = fmaf(xv.w, wc[(k + 3) * 32], acc);
            }
            acc = fmaxf(acc, 0.f);
            const int col = blk * 32 + hc;
            h2buf[hb * kHID + col] = acc;
            Hbf[((size_t)t * kB + hb) * kHID + col] = bf16r(acc);
        }
        gbar(bar);
        for (int idx = j; idx < kB * kHID / 4; idx += 256)
            ((float4*)xs)[idx] = ((const float4*)h2buf)[idx];
        __syncthreads();
        // ---- ctx_raw slice = h2 @ Wuslice + bu ----
        if (j < kB * 16) {
            const int b = j >> 4, c = j & 15;
            float acc = buv;
            const float* xr = xs + b * kHID;
            const float* wc = wus + c;
            #pragma unroll 4
            for (int k = 0; k < kHID; k += 4) {
                const float4 xv = *(const float4*)(xr + k);
                acc = fmaf(xv.x, wc[(k + 0) * 16], acc);
                acc = fmaf(xv.y, wc[(k + 1) * 16], acc);
                acc = fmaf(xv.z, wc[(k + 2) * 16], acc);
                acc = fmaf(xv.w, wc[(k + 3) * 16], acc);
            }
            ctx_raw[b * kCTX + blk * 16 + c] = acc;
        }
        gbar(bar);
    }
    // final traj row (t = kS-1)
    if (blk == 0) {
        const LN8 o = ln_row(ctx_raw + hb * kCTX, l, cg0, cg1, cb0, cb1);
        *(float4*)(traj + ((size_t)hb * kS + (kS - 1)) * kCTX + l * 8)     = o.a;
        *(float4*)(traj + ((size_t)hb * kS + (kS - 1)) * kCTX + l * 8 + 4) = o.b;
    }
}

// ---------------------------------------------------------------------------
// Phase 2: logits[M=4096, N=32000] = H @ Wob + bo (Wob is [N][K] bf16).
// Block tile 128x64, 8 waves, wave tile 16x64 (4x v8f accumulators).
// k-loop fully unrolled: accumulators pinned, in-place wmma accumulate.
// LDS strides = 40 halves (80B): bank-conflict-free, b128-aligned fragments.
// ---------------------------------------------------------------------------
__global__ void __launch_bounds__(256, 2) k_logits(
    const unsigned short* __restrict__ A,    // Hbf [4096][512] row-major (m = t*8+b)
    const unsigned short* __restrict__ Bw,   // Wob [32000][512] row-major (n-major!)
    const float* __restrict__ bo,
    float* __restrict__ out) {
    __shared__ unsigned short As[128 * 40];  // [m][k-chunk 32]
    __shared__ unsigned short Bs[64 * 40];   // [n][k-chunk 32]
    const int j = threadIdx.x;
    const int lane = j & 31, wv = j >> 5;
    const int n0 = blockIdx.x * 64;
    const int m0 = blockIdx.y * 128;

    v8f acc[4] = {};

    // cooperative-load indices
    const int ra = j >> 1, ka = (j & 1) * 16;            // A: 16 halves/thread (128 rows)
    const size_t a_g = (size_t)(m0 + ra) * kHID + ka;
    const int rb = j >> 2, kb = (j & 3) * 8;             // B: 8 halves/thread (64 rows)
    const size_t b_g = (size_t)(n0 + rb) * kHID + kb;

    // fragment geometry (ISA 7.12.2: 16-bit A 16x32 and B 32x16 lane layouts)
    const int m_f = lane & 15, hi8 = (lane >> 4) * 8;
    const unsigned short* a_base = As + (wv * 16 + m_f) * 40 + hi8;
    const int n_f = lane & 15, kg16 = (lane >> 4) * 8;

    union UF { uint4 u[2]; v16bf v; };

    #pragma unroll
    for (int kt = 0; kt < 16; ++kt) {
        const int k0 = kt * 32;
        // global -> regs (no LDS touch yet)
        const uint4* ga = (const uint4*)(A + a_g + k0);
        const uint4 a0 = ga[0], a1 = ga[1];
        const uint4 b0 = *(const uint4*)(Bw + b_g + k0);
        if (kt < 15) {
            __builtin_prefetch((const void*)(A + a_g + k0 + 32), 0, 1);
            __builtin_prefetch((const void*)(Bw + b_g + k0 + 32), 0, 1);
        }
        __syncthreads();                    // previous iteration's readers done
        *(uint4*)(As + ra * 40 + ka)     = a0;
        *(uint4*)(As + ra * 40 + ka + 8) = a1;
        *(uint4*)(Bs + rb * 40 + kb)     = b0;
        __syncthreads();
        // A fragment: halves [hi8, hi8+8) and [16+hi8, 16+hi8+8) of row m
        UF af;
        af.u[0] = *(const uint4*)(a_base);
        af.u[1] = *(const uint4*)(a_base + 16);
        #pragma unroll
        for (int nt = 0; nt < 4; ++nt) {
            // B fragment: column n = nt*16+n_f, halves [kg16, kg16+8) and [16+kg16, 16+kg16+8)
            const unsigned short* bb = Bs + (nt * 16 + n_f) * 40 + kg16;
            UF bf_;
            bf_.u[0] = *(const uint4*)(bb);
            bf_.u[1] = *(const uint4*)(bb + 16);
            acc[nt] = __builtin_amdgcn_wmma_f32_16x16x32_bf16(
                false, af.v, false, bf_.v, (short)0, acc[nt], false, false);
        }
    }

    // C/D layout: VGPR r, lanes<16 -> M=r, lanes>=16 -> M=r+8; N = lane%16.
    // Nontemporal: the 524MB logits stream must not evict L2-resident H/Wo.
    #pragma unroll
    for (int nt = 0; nt < 4; ++nt) {
        const int n_g = n0 + nt * 16 + n_f;
        const float biasv = bo[n_g];
        #pragma unroll
        for (int rr = 0; rr < 8; ++rr) {
            const int m = m0 + wv * 16 + (lane >> 4) * 8 + rr;  // m = t*8 + b
            const int tt = m >> 3, b = m & 7;
            __builtin_nontemporal_store(acc[nt][rr] + biasv,
                                        out + ((size_t)b * kS + tt) * kV + n_g);
        }
    }
}

// ---------------------------------------------------------------------------
extern "C" void kernel_launch(void* const* d_in, const int* in_sizes, int n_in,
                              void* d_out, int out_size, void* d_ws, size_t ws_size,
                              hipStream_t stream) {
    const int*   ids = (const int*)d_in[0];
    const float* emb = (const float*)d_in[1];
    const float* eg  = (const float*)d_in[2];
    const float* eb  = (const float*)d_in[3];
    const float* W1  = (const float*)d_in[4];
    const float* b1  = (const float*)d_in[5];
    const float* W2  = (const float*)d_in[6];
    const float* b2  = (const float*)d_in[7];
    const float* Wo  = (const float*)d_in[8];
    const float* bo  = (const float*)d_in[9];
    const float* Wu  = (const float*)d_in[10];
    const float* bu  = (const float*)d_in[11];
    const float* cng = (const float*)d_in[12];
    const float* cnb = (const float*)d_in[13];

    char* ws = (char*)d_ws;   // needs ~41MB
    int*   bar     = (int*)(ws + OFF_BAR);
    float* ctx_raw = (float*)(ws + OFF_CTXRAW);
    float* hbuf    = (float*)(ws + OFF_HBUF);
    float* h2buf   = (float*)(ws + OFF_H2);
    float* tok     = (float*)(ws + OFF_TOK);
    unsigned short* Hbf = (unsigned short*)(ws + OFF_HBF);
    unsigned short* Wob = (unsigned short*)(ws + OFF_WOBF);

    float* logits = (float*)d_out;
    float* traj   = logits + (size_t)kB * kS * kV;

    k_embed<<<kB * kS, 256, 0, stream>>>(ids, emb, eg, eb, tok, bar);
    k_cvt_wo<<<dim3(kV / 32, kHID / 32), 256, 0, stream>>>(Wo, Wob);
    const size_t lds1 = (size_t)(kKTOT * 32 + kHID * 32 + kHID * 16 + kB * kKTOT) * sizeof(float); // 176KB
    k_scan<<<kNBLK, 256, lds1, stream>>>(W1, b1, W2, b2, Wu, bu, cng, cnb, tok,
                                         ctx_raw, hbuf, h2buf, Hbf, traj, bar);
    k_logits<<<dim3(kV / 64, (kB * kS) / 128), 256, 0, stream>>>(Hbf, Wob, bo, logits);
}
